// ProtoVAE_tab_22204980920725
// MI455X (gfx1250) — matrix operand order repaired and checked by reading, not compiled, not run on hardware
//
#include <hip/hip_runtime.h>
#include <hip/hip_bf16.h>
#include <math.h>

#define BATCH   16384
#define INSZ    512
#define LAT     256
#define NPROTO  128
#define NCLASS  8
#define PC      16
#define EPSV    1e-4f
#define LOGMIN  -18.420680743952367f

typedef __attribute__((ext_vector_type(16))) __bf16 v16bf;
typedef __attribute__((ext_vector_type(8)))  float  v8f;
typedef __attribute__((ext_vector_type(4)))  unsigned int v4u;
typedef __attribute__((ext_vector_type(8)))  int  v8i;
typedef __attribute__((ext_vector_type(4)))  int  v4i;

union Frag { uint4 q[2]; v16bf v; };

// f32 -> bf16 round-to-nearest-even, bit-level
__device__ __forceinline__ unsigned short f2bf(float f) {
  unsigned u = __builtin_bit_cast(unsigned, f);
  u += 0x7FFFu + ((u >> 16) & 1u);
  return (unsigned short)(u >> 16);
}
__device__ __forceinline__ unsigned pk2(float a, float b) {
  return (unsigned)f2bf(a) | ((unsigned)f2bf(b) << 16);
}

// ---------------------------------------------------------------------------
// Tensor Data Mover: DMA one 128-row x 64-col bf16 tile (row pitch K elems)
// from global into LDS at lds_off, with LDS padding of 16B every 128B so the
// staged tile has a 144B row pitch (bank-conflict-free b128 fragment reads).
// D# layout per CDNA5 ISA ch.8; 6-arg clang-23 builtin form
//   (g0 u32x4, g1 i32x8, g2 i32x4, g3 i32x4, i32x8, cpol).
// ---------------------------------------------------------------------------
__device__ __forceinline__ void tdm_load_tile_128x64(unsigned lds_off,
                                                     const unsigned short* gptr,
                                                     int K)
{
  const unsigned long long ga = (unsigned long long)(size_t)gptr;
  v4u g0;
  g0[0] = 1u;                                   // count=1 (valid user D#)
  g0[1] = lds_off;                              // lds_addr (bytes)
  g0[2] = (unsigned)(ga & 0xFFFFFFFFu);         // global_addr[31:0]
  g0[3] = (unsigned)((ga >> 32) & 0x01FFFFFFu)  // global_addr[56:32]
        | (2u << 30);                           // type=2 ("image")
  v8i g1;
  g1[0] = (int)((1u << 16)                      // data_size = 1 -> 2 bytes
              | (1u << 20)                      // pad_enable
              | (4u << 22)                      // pad_interval: 32 dwords (128B)
              | (3u << 25));                    // pad_amount: 4 dwords (16B)
  g1[1] = (int)(((unsigned)K & 0xFFFFu) << 16);         // tensor_dim0[15:0]
  g1[2] = (int)((((unsigned)K >> 16) & 0xFFFFu)         // tensor_dim0[31:16]
              | (128u << 16));                          // tensor_dim1[15:0]=128
  g1[3] = (int)(64u << 16);                             // tile_dim0=64 [127:112]
  g1[4] = (int)(128u);                                  // tile_dim1=128; tile_dim2=0
  g1[5] = (int)((unsigned)K);                           // tensor_dim0_stride[31:0]
  g1[6] = 0;                                            // stride[47:32]=0, dim1_stride=0
  g1[7] = 0;
  const v4i g2 = {0, 0, 0, 0};                  // 2-D tensor: groups 2/3 unused
  const v4i g3 = {0, 0, 0, 0};
  const v8i gx = {0, 0, 0, 0, 0, 0, 0, 0};      // extra group (unused)
  __builtin_amdgcn_tensor_load_to_lds(g0, g1, g2, g3, gx, 0);
}

// ---------------------------------------------------------------------------
// f32 -> bf16 bulk convert (n must be a multiple of 1024; grid = n/1024)
// ---------------------------------------------------------------------------
__global__ __launch_bounds__(256)
void cvt_bf16_kernel(const float* __restrict__ in, unsigned short* __restrict__ out)
{
  const int idx = blockIdx.x * 256 + threadIdx.x;     // float4 index
  const float4 v = ((const float4*)in)[idx];
  ((uint2*)out)[idx] = make_uint2(pk2(v.x, v.y), pk2(v.z, v.w));
}

// ---------------------------------------------------------------------------
// Tiled bf16-WMMA GEMM with TDM double-buffered staging:
//   out = epilogue(A[M,K]_bf16 @ W[N,K]_bf16^T + bias[N])
// Workgroup tile 128x128, 8 waves (4M x 2N), wave tile 32x64 (2x4 frags).
// EPI: 0 = ReLU   -> bf16 outb[M,N]
//      1 = split  -> mu: f32 outf0[M,256] + bf16 outb[M,256]; lv: clip -> outf1
//      2 = tanh   -> f32 outf0[M,N]
// ---------------------------------------------------------------------------
template<int EPI>
__global__ __launch_bounds__(256)
void gemm_bf16_wmma(const unsigned short* __restrict__ A,
                    const unsigned short* __restrict__ W,
                    const float* __restrict__ bias,
                    float* __restrict__ outf0, float* __restrict__ outf1,
                    unsigned short* __restrict__ outb,
                    int M, int N, int K)
{
  __shared__ unsigned short As[2][128 * 72];    // 144B row pitch (72 ushorts)
  __shared__ unsigned short Bs[2][128 * 72];

  const int tid  = threadIdx.x;
  const int lane = tid & 31;
  const int wave = tid >> 5;
  const int wm   = wave & 3;
  const int wn   = wave >> 2;
  const int hf   = lane >> 4;
  const int l16  = lane & 15;
  const int m0   = blockIdx.x * 128;
  const int n0   = blockIdx.y * 128;

  // Unprovable dummy store: keeps LLVM from treating the TDM-written LDS
  // arrays as never-stored (M is a runtime argument, never 0 here).
  if (__builtin_expect(M == 0, 0)) { As[0][0] = (unsigned short)N; Bs[0][0] = (unsigned short)K; }

  const unsigned ldsA0 = (unsigned)(size_t)&As[0][0];
  const unsigned ldsA1 = (unsigned)(size_t)&As[1][0];
  const unsigned ldsB0 = (unsigned)(size_t)&Bs[0][0];
  const unsigned ldsB1 = (unsigned)(size_t)&Bs[1][0];

  v8f acc[2][4];
  #pragma unroll
  for (int i = 0; i < 2; ++i)
    #pragma unroll
    for (int j = 0; j < 4; ++j)
      #pragma unroll
      for (int r = 0; r < 8; ++r) acc[i][j][r] = 0.0f;

  const int S = K >> 6;                          // stages of 64 K-elems
  if (wave == 0) {                               // prologue: stage 0 -> buffer 0
    tdm_load_tile_128x64(ldsA0, A + (size_t)m0 * K, K);
    tdm_load_tile_128x64(ldsB0, W + (size_t)n0 * K, K);
  }

  for (int s = 0; s < S; ++s) {
    if (wave == 0) __builtin_amdgcn_s_wait_tensorcnt(0);
    __syncthreads();                             // stage-s tiles visible to all
    if (wave == 0 && (s + 1) < S) {              // overlap: DMA stage s+1
      const int k1 = (s + 1) << 6;
      tdm_load_tile_128x64(((s + 1) & 1) ? ldsA1 : ldsA0,
                           A + (size_t)m0 * K + k1, K);
      tdm_load_tile_128x64(((s + 1) & 1) ? ldsB1 : ldsB0,
                           W + (size_t)n0 * K + k1, K);
    }

    const unsigned short* Asb = As[s & 1];
    const unsigned short* Bsb = Bs[s & 1];
    #pragma unroll
    for (int ks = 0; ks < 64; ks += 32) {
      Frag a[2], b[4];
      // A 16x32 bf16 fragment: VGPR0-3 -> K(0..7)+8*hf, VGPR4-7 -> +16
      #pragma unroll
      for (int i = 0; i < 2; ++i) {
        const unsigned short* base = &Asb[(wm * 32 + i * 16 + l16) * 72 + ks + 8 * hf];
        a[i].q[0] = *(const uint4*)(base);
        a[i].q[1] = *(const uint4*)(base + 16);
      }
      // B 32x16 bf16 fragment: VGPR0-3 -> K(0..7)+8*hf, VGPR4-7 -> +16
      #pragma unroll
      for (int j = 0; j < 4; ++j) {
        const unsigned short* base = &Bsb[(wn * 64 + j * 16 + l16) * 72 + ks + 8 * hf];
        b[j].q[0] = *(const uint4*)(base);
        b[j].q[1] = *(const uint4*)(base + 16);
      }
      #pragma unroll
      for (int i = 0; i < 2; ++i)
        #pragma unroll
        for (int j = 0; j < 4; ++j)
          acc[i][j] = __builtin_amdgcn_wmma_f32_16x16x32_bf16(
              false, a[i].v, false, b[j].v, (short)0, acc[i][j], false, false);
    }
    __syncthreads();                             // done reading buffer s&1
  }

  // Epilogue. C/D layout: VGPR r -> M = r + 8*hf, N = lane&15.
  #pragma unroll
  for (int i = 0; i < 2; ++i) {
    #pragma unroll
    for (int j = 0; j < 4; ++j) {
      const int col = n0 + wn * 64 + j * 16 + l16;
      const float bval = bias[col];
      #pragma unroll
      for (int r = 0; r < 8; ++r) {
        const int row = m0 + wm * 32 + i * 16 + 8 * hf + r;
        float v = acc[i][j][r] + bval;
        if (EPI == 0) {                          // ReLU -> bf16 activation
          v = fmaxf(v, 0.0f);
          outb[(size_t)row * N + col] = f2bf(v);
        } else if (EPI == 2) {                   // tanh -> f32 final output
          outf0[(size_t)row * N + col] = tanhf(v);
        } else {                                 // split mu / clipped logVar
          if (col < LAT) {
            outf0[(size_t)row * LAT + col] = v;          // z (f32, for proto)
            outb[(size_t)row * LAT + col] = f2bf(v);     // z (bf16, for GEMM3)
          } else {
            v = fminf(fmaxf(v, LOGMIN), -LOGMIN);
            outf1[(size_t)row * LAT + (col - LAT)] = v;  // logVar (f32)
          }
        }
      }
    }
  }
}

// ---------------------------------------------------------------------------
// Per-sample prototype kernel: one wave32 per sample.
// out[b] = sum_j sim_j * W_last[t*16+j];  ratios[b] = num/den.
// kl_j = (sum_l 0.5*(var - lv - 1) + 0.5*d2_j) / 256
// ---------------------------------------------------------------------------
__global__ __launch_bounds__(256)
void proto_kernel(const float* __restrict__ z, const float* __restrict__ lv,
                  const int* __restrict__ t, const float* __restrict__ protos,
                  const float* __restrict__ Wlast,
                  float* __restrict__ out, float* __restrict__ ratios)
{
  const int lane = threadIdx.x & 31;
  const int wave = threadIdx.x >> 5;
  const int s = blockIdx.x * 8 + wave;

  const float* zrow  = z  + (size_t)s * LAT + lane * 8;
  const float* lvrow = lv + (size_t)s * LAT + lane * 8;
  const float4 z0 = *(const float4*)(zrow);
  const float4 z1 = *(const float4*)(zrow + 4);
  const float4 l0 = *(const float4*)(lvrow);
  const float4 l1 = *(const float4*)(lvrow + 4);
  const float zz[8]  = { z0.x, z0.y, z0.z, z0.w, z1.x, z1.y, z1.z, z1.w };
  const float lvv[8] = { l0.x, l0.y, l0.z, l0.w, l1.x, l1.y, l1.z, l1.w };

  float s0 = 0.0f;                        // sum_l 0.5*(var - lv - 1)
  #pragma unroll
  for (int e = 0; e < 8; ++e) s0 += 0.5f * (expf(lvv[e]) - lvv[e] - 1.0f);
  #pragma unroll
  for (int m = 16; m >= 1; m >>= 1) s0 += __shfl_xor(s0, m);

  const int cls = t[s];
  float num = 0.0f, den = 0.0f, ov = 0.0f;

  #pragma unroll 4
  for (int j = 0; j < PC; ++j) {
    const float* pr = protos + ((size_t)(cls * PC + j)) * LAT + lane * 8;
    const float4 p0 = *(const float4*)(pr);
    const float4 p1 = *(const float4*)(pr + 4);
    const float pp[8] = { p0.x, p0.y, p0.z, p0.w, p1.x, p1.y, p1.z, p1.w };
    float d2 = 0.0f;
    #pragma unroll
    for (int e = 0; e < 8; ++e) { const float df = zz[e] - pp[e]; d2 += df * df; }
    #pragma unroll
    for (int m = 16; m >= 1; m >>= 1) d2 += __shfl_xor(d2, m);

    const float d   = sqrtf(d2);
    const float sim = logf((d + 1.0f) / (d + EPSV));
    const float kl  = (s0 + 0.5f * d2) * (1.0f / (float)LAT);
    const float klw = kl * sim;
    num += klw;
    den += (klw > 0.0f) ? sim : 0.0f;
    ov  += sim * Wlast[cls * PC + j];
  }
  if (lane == 0) {
    out[s]    = ov;
    ratios[s] = num / den;
  }
}

// ---------------------------------------------------------------------------
// Per-class Gram orthogonality: one block per class -> norms[c]
// ---------------------------------------------------------------------------
__global__ __launch_bounds__(256)
void ortho_kernel(const float* __restrict__ protos, float* __restrict__ norms)
{
  __shared__ float pc[PC][LAT];
  __shared__ float red[256];
  const int c = blockIdx.x;
  const int tid = threadIdx.x;            // tid == latent index l

  float vals[PC];
  float mean = 0.0f;
  #pragma unroll
  for (int i = 0; i < PC; ++i) {
    vals[i] = protos[((size_t)c * PC + i) * LAT + tid];
    mean += vals[i];
  }
  mean *= (1.0f / (float)PC);
  #pragma unroll
  for (int i = 0; i < PC; ++i) pc[i][tid] = vals[i] - mean;
  __syncthreads();

  const int gi = tid >> 4, gj = tid & 15;
  float g = 0.0f;
  for (int l = 0; l < LAT; ++l) g += pc[gi][l] * pc[gj][l];
  if (gi == gj) g -= 1.0f;
  red[tid] = g * g;
  __syncthreads();
  for (int st = 128; st >= 1; st >>= 1) {
    if (tid < st) red[tid] += red[tid + st];
    __syncthreads();
  }
  if (tid == 0) norms[c] = sqrtf(red[0]);
}

// ---------------------------------------------------------------------------
// Deterministic fixed-order final reductions -> kl_loss, ortho_loss
// ---------------------------------------------------------------------------
__global__ __launch_bounds__(256)
void finalize_kernel(const float* __restrict__ ratios, const float* __restrict__ norms,
                     float* __restrict__ kl_out, float* __restrict__ ortho_out)
{
  __shared__ float red[256];
  const int tid = threadIdx.x;
  float s = 0.0f;
  for (int k = 0; k < BATCH / 256; ++k) s += ratios[tid + k * 256];
  red[tid] = s;
  __syncthreads();
  for (int st = 128; st >= 1; st >>= 1) {
    if (tid < st) red[tid] += red[tid + st];
    __syncthreads();
  }
  if (tid == 0) {
    *kl_out = red[0] * (1.0f / (float)BATCH);
    float o = 0.0f;
    for (int c = 0; c < NCLASS; ++c) o += norms[c];
    *ortho_out = o * (1.0f / (float)NCLASS);
  }
}

// ---------------------------------------------------------------------------
extern "C" void kernel_launch(void* const* d_in, const int* in_sizes, int n_in,
                              void* d_out, int out_size, void* d_ws, size_t ws_size,
                              hipStream_t stream)
{
  const float* x    = (const float*)d_in[0];
  const int*   t    = (const int*)  d_in[1];
  const float* prot = (const float*)d_in[2];
  const float* W1   = (const float*)d_in[3];
  const float* b1   = (const float*)d_in[4];
  const float* W2   = (const float*)d_in[5];
  const float* b2   = (const float*)d_in[6];
  const float* Wd1  = (const float*)d_in[7];
  const float* bd1  = (const float*)d_in[8];
  const float* Wd2  = (const float*)d_in[9];
  const float* bd2  = (const float*)d_in[10];
  const float* Wl   = (const float*)d_in[11];

  float* out = (float*)d_out;                               // [BATCH]
  float* dec = out + BATCH;                                 // [BATCH, 512]
  float* klp = out + BATCH + (size_t)BATCH * INSZ;          // scalar
  float* orp = klp + 1;                                     // scalar

  // ---- workspace layout ----
  unsigned short* wsb = (unsigned short*)d_ws;
  unsigned short* x_bf   = wsb;                              // 8,388,608
  unsigned short* h_bf   = x_bf + (size_t)BATCH * INSZ;      // 8,388,608
  unsigned short* dh_bf  = h_bf;                             // alias (h dead)
  unsigned short* z_bf   = x_bf;                             // alias (x dead)
  unsigned short* w1_bf  = h_bf  + (size_t)BATCH * INSZ;     // 262,144
  unsigned short* w2_bf  = w1_bf + (size_t)INSZ * INSZ;      // 262,144
  unsigned short* wd1_bf = w2_bf + (size_t)INSZ * INSZ;      // 65,536
  unsigned short* wd2_bf = wd1_bf + (size_t)LAT * LAT;       // 131,072
  float* fbase  = (float*)(wd2_bf + (size_t)INSZ * LAT);
  float* z_f32  = fbase;                                     // 4,194,304
  float* lv_f32 = z_f32 + (size_t)BATCH * LAT;               // 4,194,304
  float* ratios = lv_f32 + (size_t)BATCH * LAT;              // 16,384
  float* norms  = ratios + BATCH;                            // 8

  const dim3 blk(256);

  // ---- one-time f32 -> bf16 conversions (x_tab + weights) ----
  cvt_bf16_kernel<<<dim3((BATCH * INSZ) / 1024), blk, 0, stream>>>(x, x_bf);
  cvt_bf16_kernel<<<dim3((INSZ * INSZ) / 1024), blk, 0, stream>>>(W1, w1_bf);
  cvt_bf16_kernel<<<dim3((INSZ * INSZ) / 1024), blk, 0, stream>>>(W2, w2_bf);
  cvt_bf16_kernel<<<dim3((LAT * LAT) / 1024), blk, 0, stream>>>(Wd1, wd1_bf);
  cvt_bf16_kernel<<<dim3((INSZ * LAT) / 1024), blk, 0, stream>>>(Wd2, wd2_bf);

  // h = relu(x @ W1^T + b1)            [16384 x 512] -> bf16
  gemm_bf16_wmma<0><<<dim3(BATCH / 128, INSZ / 128), blk, 0, stream>>>(
      x_bf, w1_bf, b1, nullptr, nullptr, h_bf, BATCH, INSZ, INSZ);
  // conv = h @ W2^T + b2 -> mu (f32+bf16) / clip(logVar) (f32)
  gemm_bf16_wmma<1><<<dim3(BATCH / 128, INSZ / 128), blk, 0, stream>>>(
      h_bf, w2_bf, b2, z_f32, lv_f32, z_bf, BATCH, INSZ, INSZ);
  // prototype similarities, out[b], per-sample KL ratios
  proto_kernel<<<dim3(BATCH / 8), blk, 0, stream>>>(
      z_f32, lv_f32, t, prot, Wl, out, ratios);
  // dh = relu(z @ Wd1^T + bd1)         [16384 x 256] -> bf16 (aliases h_bf)
  gemm_bf16_wmma<0><<<dim3(BATCH / 128, LAT / 128), blk, 0, stream>>>(
      z_bf, wd1_bf, bd1, nullptr, nullptr, dh_bf, BATCH, LAT, LAT);
  // decoded = tanh(dh @ Wd2^T + bd2)   [16384 x 512] -> f32 d_out
  gemm_bf16_wmma<2><<<dim3(BATCH / 128, INSZ / 128), blk, 0, stream>>>(
      dh_bf, wd2_bf, bd2, dec, nullptr, nullptr, BATCH, INSZ, LAT);
  // per-class Gram norms, then deterministic scalar reductions
  ortho_kernel<<<dim3(NCLASS), blk, 0, stream>>>(prot, norms);
  finalize_kernel<<<dim3(1), blk, 0, stream>>>(ratios, norms, klp, orp);
}
